// MSU_69200513073591
// MI455X (gfx1250) — matrix operand order repaired and checked by reading, not compiled, not run on hardware
//
#include <hip/hip_runtime.h>
#include <hip/hip_bf16.h>
#include <cmath>

// ---------------- problem constants ----------------
#define BB    2
#define SS    512
#define D_IN  1024
#define D_OUT 1024
#define HH    8
#define DK    64
#define DV    128
#define KC    4               // conv kernel width
#define INTER 2560            // H*(DK+DK+DV+DK)
#define GATE  1024            // H*DV
#define PROJN 3584            // INTER+GATE
#define EPSV  1e-6f

typedef __attribute__((ext_vector_type(16))) __bf16 v16bf;
typedef __attribute__((ext_vector_type(8)))  float  v8f;

union FragBF {
  v16bf v;
  unsigned short s[16];
  unsigned u[8];
  uint4 q[2];
};

// packed f32x2 -> bf16x2 (RNE): native gfx1250 v_cvt_pk_bf16_f32 via inline asm
__device__ __forceinline__ unsigned pk2bf(float a, float b) {
  unsigned r;
  asm volatile("v_cvt_pk_bf16_f32 %0, %1, %2" : "=v"(r) : "v"(a), "v"(b));
  return r;
}

__device__ __forceinline__ unsigned short f2bf(float f) {
  return (unsigned short)(pk2bf(f, f) & 0xFFFFu);
}

__device__ __forceinline__ float sigmoidf_(float x) { return 1.f / (1.f + expf(-x)); }

// native gfx1250 transcendental tanh
__device__ __forceinline__ float fast_tanh(float x) {
  float r;
  asm volatile("v_tanh_f32 %0, %1" : "=v"(r) : "v"(x));
  return r;
}

// ----- A-fragment (16x32, 16-bit): halves hold K {0-7,16-23} / {8-15,24-31} -----
__device__ __forceinline__ v16bf load_fragA_f32(const float* p, int ld, int row0, int kc) {
  int lane = threadIdx.x & 31;
  int r = lane & 15;
  int half8 = (lane >> 4) << 3;          // 0 or 8
  const float* b0 = p + (size_t)(row0 + r) * ld + kc + half8;
  FragBF f;
  float4 x0 = *(const float4*)(b0);
  float4 x1 = *(const float4*)(b0 + 4);
  float4 y0 = *(const float4*)(b0 + 16);
  float4 y1 = *(const float4*)(b0 + 20);
  f.u[0] = pk2bf(x0.x, x0.y); f.u[1] = pk2bf(x0.z, x0.w);
  f.u[2] = pk2bf(x1.x, x1.y); f.u[3] = pk2bf(x1.z, x1.w);
  f.u[4] = pk2bf(y0.x, y0.y); f.u[5] = pk2bf(y0.z, y0.w);
  f.u[6] = pk2bf(y1.x, y1.y); f.u[7] = pk2bf(y1.z, y1.w);
  return f.v;
}

// ----- B-fragment (32x16, 16-bit): halves hold K {0-7,16-23} / {8-15,24-31} -----
// source is B^T row-major: row = N index, contiguous in K.
__device__ __forceinline__ v16bf load_fragB_f32(const float* p, int ld, int row0, int kc) {
  int lane = threadIdx.x & 31;
  int r = lane & 15;
  int half16 = (lane >> 4) << 3;         // 0 or 8
  const float* b0 = p + (size_t)(row0 + r) * ld + kc + half16;
  FragBF f;
#pragma unroll
  for (int i = 0; i < 4; ++i) {
    float4 x = *(const float4*)(b0 + (i & 1) * 4 + (i >> 1) * 16);
    f.u[i*2+0] = pk2bf(x.x, x.y);
    f.u[i*2+1] = pk2bf(x.z, x.w);
  }
  return f.v;
}

// LDS bf16 variants (bits already bf16)
__device__ __forceinline__ v16bf load_fragA_lds(const unsigned short* p, int ld, int row0, int kc) {
  int lane = threadIdx.x & 31;
  int r = lane & 15;
  int half8 = (lane >> 4) << 3;
  const unsigned short* b = p + (row0 + r) * ld + kc + half8;
  FragBF f;
  f.q[0] = *(const uint4*)(b);        // K +0..7
  f.q[1] = *(const uint4*)(b + 16);   // K +16..23
  return f.v;
}
__device__ __forceinline__ v16bf load_fragB_lds(const unsigned short* p, int ld, int row0, int kc) {
  int lane = threadIdx.x & 31;
  int r = lane & 15;
  int half16 = (lane >> 4) << 3;
  const unsigned short* b = p + (row0 + r) * ld + kc + half16;
  FragBF f;
  f.q[0] = *(const uint4*)(b);        // K +0..7
  f.q[1] = *(const uint4*)(b + 16);   // K +16..23
  return f.v;
}

// ----- store 16x16 f32 C tile: lane = N, vgpr = M (+8 for upper half-wave) -----
__device__ __forceinline__ void store_tile(float* p, int ld, int m0, int n0, v8f c) {
  int lane = threadIdx.x & 31;
  int n = lane & 15;
  int moff = (lane >> 4) * 8;
#pragma unroll
  for (int r = 0; r < 8; ++r)
    p[(size_t)(m0 + moff + r) * ld + n0 + n] = c[r];
}

// =======================================================================
// Kernel 1/6: generic bf16-WMMA GEMM  C(M,N) = A(M,K) * Bt(N,K)^T
// block = 128 (4 waves); wave computes 32(M) x 64(N): 2 A-frags reused
// across 4 B-frags -> 8 WMMAs per K-chunk.  Requires M % 128 == 0.
// =======================================================================
__global__ __launch_bounds__(128) void gemm_bf16_kernel(
    const float* __restrict__ A, const float* __restrict__ Bt, float* __restrict__ C,
    int Kd, int lda, int ldb, int ldc) {
  int wave = threadIdx.x >> 5;
  int m0 = (blockIdx.y * 8 + wave * 2) * 16;   // two m-tiles per wave
  int n0 = blockIdx.x * 64;
  v8f acc[2][4];
#pragma unroll
  for (int i = 0; i < 2; ++i)
#pragma unroll
    for (int j = 0; j < 4; ++j)
#pragma unroll
      for (int r = 0; r < 8; ++r) acc[i][j][r] = 0.f;

  for (int kc = 0; kc < Kd; kc += 32) {
    v16bf a0 = load_fragA_f32(A, lda, m0, kc);
    v16bf a1 = load_fragA_f32(A, lda, m0 + 16, kc);
#pragma unroll
    for (int j = 0; j < 4; ++j) {
      v16bf b = load_fragB_f32(Bt, ldb, n0 + j * 16, kc);
      acc[0][j] = __builtin_amdgcn_wmma_f32_16x16x32_bf16(false, a0, false, b, (short)0, acc[0][j], false, false);
      acc[1][j] = __builtin_amdgcn_wmma_f32_16x16x32_bf16(false, a1, false, b, (short)0, acc[1][j], false, false);
    }
  }
#pragma unroll
  for (int i = 0; i < 2; ++i)
#pragma unroll
    for (int j = 0; j < 4; ++j)
      store_tile(C, ldc, m0 + i * 16, n0 + j * 16, acc[i][j]);
}

// =======================================================================
// Kernel 2/6: depthwise causal conv (K=4) + SiLU on the INTER channels
// =======================================================================
__global__ __launch_bounds__(256) void conv_silu_kernel(
    const float* __restrict__ proj, const float* __restrict__ conv_w,
    float* __restrict__ y) {
  int idx = blockIdx.x * 256 + threadIdx.x;
  if (idx >= BB * SS * INTER) return;
  int c = idx % INTER;
  int s = (idx / INTER) % SS;
  int b = idx / (INTER * SS);
  float acc = 0.f;
#pragma unroll
  for (int t = 0; t < KC; ++t) {
    int sp = s - (KC - 1) + t;
    if (sp >= 0)
      acc += conv_w[c * KC + t] * proj[((size_t)(b * SS + sp)) * PROJN + c];
  }
  y[idx] = acc * sigmoidf_(acc);   // SiLU
}

// =======================================================================
// Kernel 3/6: per-(b,s,h) RMS-norms of q,k,v + forget-gate precompute
// block = 128, grid = B*S*H
// =======================================================================
__device__ __forceinline__ float block_red128(float v, float* red) {
  int tid = threadIdx.x;
  red[tid] = v;
  __syncthreads();
#pragma unroll
  for (int o = 64; o > 0; o >>= 1) {
    if (tid < o) red[tid] += red[tid + o];
    __syncthreads();
  }
  float r = red[0];
  __syncthreads();
  return r;
}

__global__ __launch_bounds__(128) void qkvf_kernel(
    const float* __restrict__ y, const float* __restrict__ lfit,
    const float* __restrict__ fbias, const float* __restrict__ log_factor,
    float* __restrict__ qn, float* __restrict__ kn,
    float* __restrict__ vn, float* __restrict__ fgb) {
  __shared__ float red[128];
  int g = blockIdx.x;                 // (b*S+s)*H + h
  int h = g % HH;
  int bs = g / HH;
  int tid = threadIdx.x;
  const float* yrow = y + (size_t)bs * INTER;

  // q
  float qv = (tid < DK) ? yrow[h * DK + tid] : 0.f;
  float sq = block_red128(qv * qv, red);
  if (tid < DK) qn[(size_t)g * DK + tid] = qv * rsqrtf(sq / DK + EPSV);
  // k
  float kv = (tid < DK) ? yrow[HH * DK + h * DK + tid] : 0.f;
  float sk = block_red128(kv * kv, red);
  if (tid < DK) kn[(size_t)g * DK + tid] = kv * rsqrtf(sk / DK + EPSV);
  // v (DV == 128 == blockDim)
  float vv = yrow[2 * HH * DK + h * DV + tid];
  float sv = block_red128(vv * vv, red);
  float factor = log1pf(expf(log_factor[h]));  // softplus
  vn[(size_t)g * DV + tid] = vv * rsqrtf(sv / DV + EPSV) * factor;
  // f -> fg = sigmoid(2*sigmoid(lfit)*(f + bias))
  if (tid < DK) {
    float fr = yrow[2 * HH * DK + HH * DV + h * DK + tid];
    float sig = sigmoidf_(lfit[h * DK + tid]);
    float fval = 2.f * sig * (fr + fbias[h * DK + tid]);
    fgb[(size_t)g * DK + tid] = sigmoidf_(fval);
  }
}

// =======================================================================
// Kernel 4/6: the sequential scan. 64 blocks x 1 wave.
// Block (b,h,kt) owns a 16x128 f32 state tile (8 WMMA C fragments).
// Per step: 32 x v_wmma_f32_16x16x32_bf16 for h@W, tanh/gate update,
// q.h partial reduced via shfl_xor(16), stored to outpart (no atomics).
// =======================================================================
__global__ __launch_bounds__(32) void scan_kernel(
    const float* __restrict__ qn, const float* __restrict__ kn,
    const float* __restrict__ vn, const float* __restrict__ fgb,
    const float* __restrict__ state_weight, const float* __restrict__ log_factor,
    float* __restrict__ outpart) {
  __shared__ __align__(16) unsigned short Wt[DV * DV];   // [w][v], bf16, 32 KB
  __shared__ __align__(16) unsigned short hA[16 * DV];   // [k_local][v], bf16, 4 KB

  int blk = blockIdx.x;            // ((b*H + h)*4 + kt)
  int kt = blk & 3;
  int h  = (blk >> 2) & 7;
  int b  = blk >> 5;
  int lane = threadIdx.x;
  int n = lane & 15;
  int moff = (lane >> 4) * 8;

  // Wt[w][v] = state_weight[h][v][w] * softplus(log_factor[h]) * sigmoid(20)
  float factor = log1pf(expf(log_factor[h]));
  const float rsig = 0.9999999979388463f;
  float wscale = factor * rsig;
  const float* Wsrc = state_weight + (size_t)h * DV * DV;
  for (int idx = lane; idx < DV * DV; idx += 32) {
    int v = idx & (DV - 1);
    int w = idx >> 7;
    Wt[w * DV + v] = f2bf(Wsrc[v * DV + w] * wscale);
  }
  __syncthreads();

  v8f Cst[8];
#pragma unroll
  for (int j = 0; j < 8; ++j)
#pragma unroll
    for (int r = 0; r < 8; ++r) Cst[j][r] = 0.f;

  for (int s = 0; s < SS; ++s) {
    // dump state tile (old h, C layout) to LDS row-major bf16
#pragma unroll
    for (int j = 0; j < 8; ++j)
#pragma unroll
      for (int r = 0; r < 8; ++r)
        hA[(moff + r) * DV + j * 16 + n] = f2bf(Cst[j][r]);

    size_t base = (size_t)(b * SS + s) * HH + h;
    const float* kp = kn  + base * DK + kt * 16;
    const float* fp = fgb + base * DK + kt * 16;
    const float* qp = qn  + base * DK + kt * 16;
    const float* vp = vn  + base * DV;

    float k_[8], f_[8], q_[8], vv[8];
#pragma unroll
    for (int r = 0; r < 8; ++r) {
      int m = moff + r;
      k_[r] = kp[m]; f_[r] = fp[m]; q_[r] = qp[m];
    }
#pragma unroll
    for (int j = 0; j < 8; ++j) vv[j] = vp[j * 16 + n];

    // Acc = k (x) v  (C layout seed), then += h @ W
    v8f Acc[8];
#pragma unroll
    for (int j = 0; j < 8; ++j)
#pragma unroll
      for (int r = 0; r < 8; ++r) Acc[j][r] = k_[r] * vv[j];

#pragma unroll
    for (int vc = 0; vc < 4; ++vc) {
      v16bf a = load_fragA_lds(hA, DV, 0, vc * 32);
#pragma unroll
      for (int j = 0; j < 8; ++j) {
        v16bf bm = load_fragB_lds(Wt, DV, j * 16, vc * 32);
        Acc[j] = __builtin_amdgcn_wmma_f32_16x16x32_bf16(false, a, false, bm, (short)0, Acc[j], false, false);
      }
    }

    // gate update + per-step output partial (sum over this tile's 16 k rows)
    float po[8];
#pragma unroll
    for (int j = 0; j < 8; ++j) {
      float sum = 0.f;
#pragma unroll
      for (int r = 0; r < 8; ++r) {
        float cand = fast_tanh(Acc[j][r]);
        float nh = f_[r] * Cst[j][r] + (1.f - f_[r]) * cand;
        Cst[j][r] = nh;
        sum += q_[r] * nh;
      }
      po[j] = sum;
    }
#pragma unroll
    for (int j = 0; j < 8; ++j) po[j] += __shfl_xor(po[j], 16, 32);

    float* op = outpart + ((size_t)blk * SS + s) * DV;
    if (lane < 16) {
#pragma unroll
      for (int j = 0; j < 4; ++j) op[j * 16 + n] = po[j];
    } else {
#pragma unroll
      for (int j = 4; j < 8; ++j) op[j * 16 + n] = po[j];
    }
  }
}

// =======================================================================
// Kernel 5/6: combine kt-partials, gate with SiLU(gate), RMS-norm (1024)
// block = 256, grid = B*S
// =======================================================================
__global__ __launch_bounds__(256) void finalize_norm_kernel(
    const float* __restrict__ outpart, const float* __restrict__ proj,
    const float* __restrict__ norm_weight, float* __restrict__ normed) {
  __shared__ float red[256];
  int bs = blockIdx.x;
  int b = bs / SS, s = bs % SS;
  int tid = threadIdx.x;
  float vals[4];
  float ss = 0.f;
#pragma unroll
  for (int i = 0; i < 4; ++i) {
    int c = tid + i * 256;         // h*DV + w
    int h = c >> 7, w = c & (DV - 1);
    float acc = 0.f;
#pragma unroll
    for (int kt = 0; kt < 4; ++kt)
      acc += outpart[(((size_t)(b * HH + h) * 4 + kt) * SS + s) * DV + w];
    float g = proj[(size_t)bs * PROJN + INTER + c];
    float t = acc * (g * sigmoidf_(g));
    vals[i] = t;
    ss += t * t;
  }
  red[tid] = ss;
  __syncthreads();
#pragma unroll
  for (int o = 128; o > 0; o >>= 1) {
    if (tid < o) red[tid] += red[tid + o];
    __syncthreads();
  }
  float rms = rsqrtf(red[0] / (float)(HH * DV) + EPSV);
#pragma unroll
  for (int i = 0; i < 4; ++i) {
    int c = tid + i * 256;
    normed[(size_t)bs * (HH * DV) + c] = vals[i] * rms * norm_weight[c];
  }
}

// =======================================================================
// launch
// =======================================================================
extern "C" void kernel_launch(void* const* d_in, const int* in_sizes, int n_in,
                              void* d_out, int out_size, void* d_ws, size_t ws_size,
                              hipStream_t stream) {
  const float* x        = (const float*)d_in[0];  // (B,S,D_IN)
  const float* w_in     = (const float*)d_in[1];  // (PROJN, D_IN)
  const float* conv_w   = (const float*)d_in[2];  // (INTER,1,K)
  const float* lfit     = (const float*)d_in[3];  // (H,DK)
  const float* fbias    = (const float*)d_in[4];  // (H,DK)
  const float* log_fac  = (const float*)d_in[5];  // (H,)
  const float* state_w  = (const float*)d_in[6];  // (H,DV,DV)
  const float* norm_w   = (const float*)d_in[7];  // (H*DV,)
  const float* w_out    = (const float*)d_in[8];  // (D_OUT, H*DV)
  float* out = (float*)d_out;

  float* ws = (float*)d_ws;
  float* proj   = ws;                                   // B*S*PROJN
  float* y      = proj  + (size_t)BB * SS * PROJN;      // B*S*INTER
  float* qn     = y     + (size_t)BB * SS * INTER;      // B*S*H*DK
  float* kn     = qn    + (size_t)BB * SS * HH * DK;
  float* fgb    = kn    + (size_t)BB * SS * HH * DK;
  float* vn     = fgb   + (size_t)BB * SS * HH * DK;    // B*S*H*DV
  float* outp   = vn    + (size_t)BB * SS * HH * DV;    // 64*S*DV
  float* normed = outp  + (size_t)64 * SS * DV;         // B*S*1024

  const int M = BB * SS;  // 1024

  // 1) proj = x @ w_in.T   (M=1024, N=3584, K=1024)
  gemm_bf16_kernel<<<dim3(PROJN / 64, M / 128), 128, 0, stream>>>(
      x, w_in, proj, D_IN, D_IN, D_IN, PROJN);

  // 2) depthwise conv + SiLU
  conv_silu_kernel<<<(BB * SS * INTER + 255) / 256, 256, 0, stream>>>(proj, conv_w, y);

  // 3) q/k/v norms + forget gates
  qkvf_kernel<<<BB * SS * HH, 128, 0, stream>>>(y, lfit, fbias, log_fac, qn, kn, vn, fgb);

  // 4) sequential scan (64 independent single-wave row-tiles)
  scan_kernel<<<64, 32, 0, stream>>>(qn, kn, vn, fgb, state_w, log_fac, outp);

  // 5) combine + gate + rmsnorm
  finalize_norm_kernel<<<BB * SS, 256, 0, stream>>>(outp, proj, norm_w, normed);

  // 6) out = normed @ w_out.T (M=1024, N=1024, K=1024)
  gemm_bf16_kernel<<<dim3(D_OUT / 64, M / 128), 128, 0, stream>>>(
      normed, w_out, out, HH * DV, HH * DV, HH * DV, D_OUT);
}
